// TransformerBlock_75488345194964
// MI455X (gfx1250) — compile-verified, matrix operand order checked by reading, not run on hardware
//
#include <hip/hip_runtime.h>

// ---------------------------------------------------------------------------
// Types for CDNA5 WMMA (gfx1250, wave32)
// ---------------------------------------------------------------------------
typedef __bf16 bf16;
typedef __attribute__((ext_vector_type(16))) __bf16 v16bf;
typedef __attribute__((ext_vector_type(8)))  __bf16 v8bf;
typedef __attribute__((ext_vector_type(8)))  float  v8f;
typedef __attribute__((ext_vector_type(4)))  int    v4i;

typedef __attribute__((address_space(1))) v4i* gv4ip;   // global int4*
typedef __attribute__((address_space(3))) v4i* lv4ip;   // LDS int4*

#define D_MODEL 1024
#define N_HEADS 16
#define D_K     64
#define HIDDEN  4096
#define SEQ     2048
#define BATCH   4
#define MTOT    (BATCH * SEQ)   // 8192 rows

__device__ __forceinline__ v8f zero8() {
  v8f z = {0.f, 0.f, 0.f, 0.f, 0.f, 0.f, 0.f, 0.f};
  return z;
}

__device__ __forceinline__ v8f wmma_bf16(v16bf a, v16bf b, v8f c) {
  // D = A(16x32 bf16) x B(32x16 bf16) + C(16x16 f32)
  return __builtin_amdgcn_wmma_f32_16x16x32_bf16(
      /*neg_a=*/false, a, /*neg_b=*/false, b,
      /*c_mod=*/(short)0, c, /*reuse_a=*/false, /*reuse_b=*/false);
}

// Async global->LDS copy of 16 bytes per lane (GLOBAL_LOAD_ASYNC_TO_LDS_B128,
// tracked with ASYNCcnt). Prototype (from toolchain diagnostic):
//   (int4 addrspace(1)*, int4 addrspace(3)*, imm offset, imm cpol)
__device__ __forceinline__ void cp_async_b128(bf16* lds_dst, const bf16* gsrc) {
  __builtin_amdgcn_global_load_async_to_lds_b128(
      (gv4ip)(v4i*)gsrc, (lv4ip)(v4i*)lds_dst, 0, 0);
}

// Generic 16x32 fragment loader from row-major [16+ rows][ld] bf16 data.
// Serves both A-fragments (lane&15 = M row) and B-fragments (lane&15 = N col,
// reading B[k][n] from a row-major [n][k] source), per ISA 7.12.2:
//   lanes 0-15: kb=0, lanes 16-31: kb=8;
//   elements 0..7  = K kb..kb+7   (contiguous)
//   elements 8..15 = K 16+kb..+7  (contiguous, +16 elements away)
union FragU { v16bf v; v8bf h[2]; };
__device__ __forceinline__ v16bf load_frag(const bf16* __restrict__ p0, int ld, int lane) {
  const bf16* p = p0 + (size_t)(lane & 15) * (size_t)ld + ((lane >> 4) << 3);
  FragU f;
  f.h[0] = *(const v8bf*)(p);
  f.h[1] = *(const v8bf*)(p + 16);
  return f.v;
}

// ---------------------------------------------------------------------------
// fp32 -> bf16 conversion (weights)
// ---------------------------------------------------------------------------
__global__ __launch_bounds__(256) void k_cvt_bf16(const float* __restrict__ in,
                                                  bf16* __restrict__ out, int n) {
  int i = (blockIdx.x * 256 + threadIdx.x) * 4;
  if (i < n) {
#pragma unroll
    for (int j = 0; j < 4; ++j) out[i + j] = (bf16)in[i + j];
  }
}

// ---------------------------------------------------------------------------
// RMSNorm: fp32 [row][1024] -> bf16, one block per row
// ---------------------------------------------------------------------------
__global__ __launch_bounds__(256) void k_rmsnorm(const float* __restrict__ x,
                                                 const float* __restrict__ g,
                                                 bf16* __restrict__ out) {
  int row = blockIdx.x;
  const float* xr = x + (size_t)row * D_MODEL;
  int base = threadIdx.x * 4;
  float v[4];
  float s = 0.f;
#pragma unroll
  for (int i = 0; i < 4; ++i) { v[i] = xr[base + i]; s += v[i] * v[i]; }
#pragma unroll
  for (int off = 16; off; off >>= 1) s += __shfl_xor(s, off, 32);
  __shared__ float ws[8];
  int lane = threadIdx.x & 31, wv = threadIdx.x >> 5;
  if (lane == 0) ws[wv] = s;
  __syncthreads();
  float tot = ws[0] + ws[1] + ws[2] + ws[3] + ws[4] + ws[5] + ws[6] + ws[7];
  float r = rsqrtf(tot * (1.f / D_MODEL) + 1e-6f);
  bf16* o = out + (size_t)row * D_MODEL + base;
#pragma unroll
  for (int i = 0; i < 4; ++i) o[i] = (bf16)(v[i] * r * g[base + i]);
}

// ---------------------------------------------------------------------------
// LayerNorm: fp32 [row][1024] -> bf16, effective scale = 1 + g
// ---------------------------------------------------------------------------
__global__ __launch_bounds__(256) void k_layernorm(const float* __restrict__ x,
                                                   const float* __restrict__ g,
                                                   const float* __restrict__ bb,
                                                   bf16* __restrict__ out) {
  int row = blockIdx.x;
  const float* xr = x + (size_t)row * D_MODEL;
  int base = threadIdx.x * 4;
  float v[4];
  float s1 = 0.f, s2 = 0.f;
#pragma unroll
  for (int i = 0; i < 4; ++i) { v[i] = xr[base + i]; s1 += v[i]; s2 += v[i] * v[i]; }
#pragma unroll
  for (int off = 16; off; off >>= 1) {
    s1 += __shfl_xor(s1, off, 32);
    s2 += __shfl_xor(s2, off, 32);
  }
  __shared__ float wa[8], wb[8];
  int lane = threadIdx.x & 31, wv = threadIdx.x >> 5;
  if (lane == 0) { wa[wv] = s1; wb[wv] = s2; }
  __syncthreads();
  float S1 = wa[0] + wa[1] + wa[2] + wa[3] + wa[4] + wa[5] + wa[6] + wa[7];
  float S2 = wb[0] + wb[1] + wb[2] + wb[3] + wb[4] + wb[5] + wb[6] + wb[7];
  float mean = S1 * (1.f / D_MODEL);
  float var  = S2 * (1.f / D_MODEL) - mean * mean;
  float r = rsqrtf(var + 1e-6f);
  bf16* o = out + (size_t)row * D_MODEL + base;
#pragma unroll
  for (int i = 0; i < 4; ++i)
    o[i] = (bf16)((v[i] - mean) * r * (1.f + g[base + i]) + bb[base + i]);
}

// ---------------------------------------------------------------------------
// WMMA GEMM: C[M,N] = A[M,K](bf16) * W[N,K]^T(bf16) + bias
// 256 threads (8 waves), 128x128 tile, 32-deep K steps.
// Tiles staged with async global->LDS copies (ASYNCcnt), double-buffered so
// the copy of step i+1 overlaps the 8 WMMAs of step i.
// Epilogues:
//  0: store f32            1: store bf16          2: relu -> bf16
//  3: per-dim q scale->bf16 (extra = per_dim_scale[64])
//  4: +extra residual -> f32 (extra = residual [M,N] f32)
//  5: transposed V store -> vT[(b*16+h)*64+d][2048] bf16
// ---------------------------------------------------------------------------
__global__ __launch_bounds__(256) void k_gemm(const bf16* __restrict__ A,
                                              const bf16* __restrict__ W,
                                              const float* __restrict__ bias,
                                              int M, int N, int K,
                                              float* __restrict__ outf,
                                              bf16* __restrict__ outb,
                                              const float* __restrict__ extra,
                                              int epi) {
  __shared__ bf16 As[2][128 * 32];
  __shared__ bf16 Bs[2][128 * 32];
  int lane = threadIdx.x & 31;
  int wv   = threadIdx.x >> 5;      // 0..7, each wave owns 16 rows x 128 cols
  int row0 = blockIdx.y * 128;
  int col0 = blockIdx.x * 128;

  v8f acc[8];
#pragma unroll
  for (int n = 0; n < 8; ++n) acc[n] = zero8();

  int tr = threadIdx.x >> 1;            // 0..127 tile row
  int tp = (threadIdx.x & 1) * 16;      // 0 or 16 (half of 32-wide K slab)
  const bf16* Ag = A + (size_t)(row0 + tr) * (size_t)K + tp;
  const bf16* Wg = W + (size_t)(col0 + tr) * (size_t)K + tp;
  bf16* Asl = &As[0][tr * 32 + tp];
  bf16* Bsl = &Bs[0][tr * 32 + tp];

  // stage one 32-deep K slab into LDS buffer `buf` (4 async b128 per thread)
  auto stage = [&](int buf, int k0) {
    cp_async_b128(Asl + buf * (128 * 32),     Ag + k0);
    cp_async_b128(Asl + buf * (128 * 32) + 8, Ag + k0 + 8);
    cp_async_b128(Bsl + buf * (128 * 32),     Wg + k0);
    cp_async_b128(Bsl + buf * (128 * 32) + 8, Wg + k0 + 8);
  };

  int nsteps = K >> 5;
  stage(0, 0);
  for (int i = 0; i < nsteps; ++i) {
    int buf = i & 1;
    if ((i + 1) < nsteps) {
      stage(buf ^ 1, (i + 1) << 5);
      __builtin_amdgcn_s_wait_asynccnt(4);  // current buffer landed; next in flight
    } else {
      __builtin_amdgcn_s_wait_asynccnt(0);  // last slab: drain everything
    }
    __syncthreads();               // all waves' copies visible
    v16bf a = load_frag(&As[buf][wv * 16 * 32], 32, lane);
#pragma unroll
    for (int n = 0; n < 8; ++n) {
      v16bf b = load_frag(&Bs[buf][n * 16 * 32], 32, lane);
      acc[n] = wmma_bf16(a, b, acc[n]);
    }
    __syncthreads();               // everyone done reading buf before overwrite
  }

  int hi = lane >> 4, nn = lane & 15;
#pragma unroll
  for (int n = 0; n < 8; ++n) {
    int gc = col0 + n * 16 + nn;
    float bs = bias[gc];
    float cs = 1.f;
    if (epi == 3) {
      float pd = extra[gc & (D_K - 1)];
      float sp = (pd > 20.f) ? pd : log1pf(__expf(pd));
      cs = 0.18033688014f * sp;   // r_softplus_0 / sqrt(64)
    }
#pragma unroll
    for (int v = 0; v < 8; ++v) {
      int gr = row0 + wv * 16 + hi * 8 + v;
      float val = acc[n][v] + bs;
      size_t idx = (size_t)gr * (size_t)N + gc;
      if (epi == 0)      outf[idx] = val;
      else if (epi == 1) outb[idx] = (bf16)val;
      else if (epi == 2) outb[idx] = (bf16)fmaxf(val, 0.f);
      else if (epi == 3) outb[idx] = (bf16)(val * cs);
      else if (epi == 4) outf[idx] = val + extra[idx];
      else {  // 5: V transposed per head: vT[(b*16+h)*64+d][s]
        int s = gr & (SEQ - 1), b = gr >> 11;
        int hh = gc >> 6, d = gc & (D_K - 1);
        outb[((size_t)((b * N_HEADS + hh) * D_K + d) << 11) + s] = (bf16)val;
      }
    }
  }
}

// ---------------------------------------------------------------------------
// Flash attention (softmax(QK^T + mask) V), one wave per 16-row Q block.
// Q,K: bf16 [B*S][1024] head-major cols; vT: bf16 [(b*16+h)*64+d][2048].
// Streams KV in 32-wide tiles; online softmax with wave32 shfl reductions.
// ---------------------------------------------------------------------------
__global__ __launch_bounds__(128) void k_attn(const bf16* __restrict__ q,
                                              const bf16* __restrict__ k,
                                              const bf16* __restrict__ vT,
                                              const float* __restrict__ mask,
                                              bf16* __restrict__ ctx) {
  __shared__ bf16 Pbuf[4][16 * 32];   // per-wave P bounce buffer
  int lane = threadIdx.x & 31;
  int wv   = threadIdx.x >> 5;
  int wid  = blockIdx.x * 4 + wv;     // 0 .. B*H*(S/16)-1 = 8191
  int qb = wid & 127;                 // query block within sequence
  int h  = (wid >> 7) & (N_HEADS - 1);
  int b  = wid >> 11;
  size_t rowbase = (size_t)b * SEQ;

  const bf16* qp = q + (rowbase + qb * 16) * D_MODEL + h * D_K;
  v16bf qa0 = load_frag(qp,      D_MODEL, lane);   // K(=d) 0..31
  v16bf qa1 = load_frag(qp + 32, D_MODEL, lane);   // K(=d) 32..63

  v8f acc[4];
#pragma unroll
  for (int t = 0; t < 4; ++t) acc[t] = zero8();
  float mrow[8], lrow[8];
#pragma unroll
  for (int i = 0; i < 8; ++i) { mrow[i] = -3.0e38f; lrow[i] = 0.f; }

  int hi = lane >> 4, nn = lane & 15;
  bf16* Pl = &Pbuf[wv][0];
  const bf16* vbase = vT + ((size_t)(b * N_HEADS + h) * D_K) * SEQ;

  for (int kv = 0; kv < SEQ; kv += 32) {
    const bf16* kp = k + (rowbase + kv) * D_MODEL + h * D_K;
    v8f s0 = zero8(), s1 = zero8();
    {
      v16bf b0 = load_frag(kp,      D_MODEL, lane);              // kv rows kv..kv+15, d 0..31
      v16bf b1 = load_frag(kp + 32, D_MODEL, lane);              // d 32..63
      s0 = wmma_bf16(qa0, b0, s0);
      s0 = wmma_bf16(qa1, b1, s0);
      v16bf b2 = load_frag(kp + 16 * D_MODEL,      D_MODEL, lane); // kv rows kv+16..kv+31
      v16bf b3 = load_frag(kp + 16 * D_MODEL + 32, D_MODEL, lane);
      s1 = wmma_bf16(qa0, b2, s1);
      s1 = wmma_bf16(qa1, b3, s1);
    }
    const float* mp = mask + (size_t)(qb * 16 + hi * 8) * SEQ + kv + nn;
#pragma unroll
    for (int v = 0; v < 8; ++v) {
      float e0 = s0[v] + mp[(size_t)v * SEQ];
      float e1 = s1[v] + mp[(size_t)v * SEQ + 16];
      float tm = fmaxf(e0, e1);
#pragma unroll
      for (int off = 1; off < 16; off <<= 1) tm = fmaxf(tm, __shfl_xor(tm, off, 32));
      float nm = fmaxf(mrow[v], tm);
      float corr = __expf(mrow[v] - nm);
      mrow[v] = nm;
      float p0 = __expf(e0 - nm);
      float p1 = __expf(e1 - nm);
      float rs = p0 + p1;
#pragma unroll
      for (int off = 1; off < 16; off <<= 1) rs += __shfl_xor(rs, off, 32);
      lrow[v] = lrow[v] * corr + rs;
#pragma unroll
      for (int t = 0; t < 4; ++t) acc[t][v] *= corr;
      // C-layout (row = hi*8+v, col = nn / 16+nn) -> row-major P in LDS
      Pl[(hi * 8 + v) * 32 + nn]      = (bf16)p0;
      Pl[(hi * 8 + v) * 32 + 16 + nn] = (bf16)p1;
    }
    asm volatile("" ::: "memory");   // keep DS writes before DS reads (HW DS is in-order)
    v16bf pa = load_frag(Pl, 32, lane);          // P as 16x32 A-fragment
#pragma unroll
    for (int t = 0; t < 4; ++t) {
      v16bf bv = load_frag(vbase + (size_t)(t * 16) * SEQ + kv, SEQ, lane); // B[k=kv][n=d]
      acc[t] = wmma_bf16(pa, bv, acc[t]);
    }
    asm volatile("" ::: "memory");
  }

#pragma unroll
  for (int v = 0; v < 8; ++v) {
    float rl = 1.f / lrow[v];
    size_t orow = (rowbase + qb * 16 + hi * 8 + v) * D_MODEL + h * D_K;
#pragma unroll
    for (int t = 0; t < 4; ++t)
      ctx[orow + t * 16 + nn] = (bf16)(acc[t][v] * rl);
  }
}

// ---------------------------------------------------------------------------
// Host-side orchestration
// ---------------------------------------------------------------------------
extern "C" void kernel_launch(void* const* d_in, const int* in_sizes, int n_in,
                              void* d_out, int out_size, void* d_ws, size_t ws_size,
                              hipStream_t stream) {
  const float* inputs = (const float*)d_in[0];
  const float* mask   = (const float*)d_in[1];
  const float* rms_g  = (const float*)d_in[2];
  const float* Wq = (const float*)d_in[3];   const float* bq = (const float*)d_in[4];
  const float* Wk = (const float*)d_in[5];   const float* bk = (const float*)d_in[6];
  const float* Wv = (const float*)d_in[7];   const float* bv = (const float*)d_in[8];
  const float* pds = (const float*)d_in[9];
  const float* Wpost = (const float*)d_in[10]; const float* bpost = (const float*)d_in[11];
  const float* ln_g = (const float*)d_in[12];  const float* ln_b = (const float*)d_in[13];
  const float* W1 = (const float*)d_in[14];  const float* b1 = (const float*)d_in[15];
  const float* W2 = (const float*)d_in[16];  const float* b2 = (const float*)d_in[17];
  float* out = (float*)d_out;

  char* wsp = (char*)d_ws;
  size_t off = 0;
  auto alloc = [&](size_t bytes) -> void* {
    void* p = wsp + off;
    off += (bytes + 255) & ~(size_t)255;
    return p;
  };

  bf16* xb    = (bf16*)alloc((size_t)MTOT * D_MODEL * 2);      // RMSNorm(x), bf16
  bf16* wqb   = (bf16*)alloc((size_t)D_MODEL * D_MODEL * 2);
  bf16* wkb   = (bf16*)alloc((size_t)D_MODEL * D_MODEL * 2);
  bf16* wvb   = (bf16*)alloc((size_t)D_MODEL * D_MODEL * 2);
  bf16* wpb   = (bf16*)alloc((size_t)D_MODEL * D_MODEL * 2);
  bf16* w1b   = (bf16*)alloc((size_t)HIDDEN * D_MODEL * 2);
  bf16* w2b   = (bf16*)alloc((size_t)D_MODEL * HIDDEN * 2);
  bf16* qbuf  = (bf16*)alloc((size_t)MTOT * D_MODEL * 2);      // scaled Q
  bf16* kbuf  = (bf16*)alloc((size_t)MTOT * D_MODEL * 2);
  bf16* vTbuf = (bf16*)alloc((size_t)MTOT * D_MODEL * 2);      // V transposed per head
  bf16* ctxb  = (bf16*)alloc((size_t)MTOT * D_MODEL * 2);      // attention output
  float* hbuf = (float*)alloc((size_t)MTOT * D_MODEL * 4);     // residual stream, f32
  bf16* lnb   = (bf16*)alloc((size_t)MTOT * D_MODEL * 2);      // LayerNorm(h)
  bf16* f1b   = (bf16*)alloc((size_t)MTOT * HIDDEN * 2);       // relu(ffn1)

  // weights -> bf16
  k_cvt_bf16<<<(D_MODEL * D_MODEL) / 1024, 256, 0, stream>>>(Wq, wqb, D_MODEL * D_MODEL);
  k_cvt_bf16<<<(D_MODEL * D_MODEL) / 1024, 256, 0, stream>>>(Wk, wkb, D_MODEL * D_MODEL);
  k_cvt_bf16<<<(D_MODEL * D_MODEL) / 1024, 256, 0, stream>>>(Wv, wvb, D_MODEL * D_MODEL);
  k_cvt_bf16<<<(D_MODEL * D_MODEL) / 1024, 256, 0, stream>>>(Wpost, wpb, D_MODEL * D_MODEL);
  k_cvt_bf16<<<(HIDDEN * D_MODEL) / 1024, 256, 0, stream>>>(W1, w1b, HIDDEN * D_MODEL);
  k_cvt_bf16<<<(HIDDEN * D_MODEL) / 1024, 256, 0, stream>>>(W2, w2b, D_MODEL * HIDDEN);

  // x = RMSNorm(inputs) -> bf16
  k_rmsnorm<<<MTOT, 256, 0, stream>>>(inputs, rms_g, xb);

  dim3 gproj(D_MODEL / 128, MTOT / 128);   // (8, 64)
  // Q (with per-dim softplus scale), K, V (transposed store)
  k_gemm<<<gproj, 256, 0, stream>>>(xb, wqb, bq, MTOT, D_MODEL, D_MODEL, nullptr, qbuf, pds, 3);
  k_gemm<<<gproj, 256, 0, stream>>>(xb, wkb, bk, MTOT, D_MODEL, D_MODEL, nullptr, kbuf, nullptr, 1);
  k_gemm<<<gproj, 256, 0, stream>>>(xb, wvb, bv, MTOT, D_MODEL, D_MODEL, nullptr, vTbuf, nullptr, 5);

  // attention: B*H*(S/16) = 8192 waves, 4 waves per block
  k_attn<<<(BATCH * N_HEADS * (SEQ / 16)) / 4, 128, 0, stream>>>(qbuf, kbuf, vTbuf, mask, ctxb);

  // post projection + residual -> h (f32)
  k_gemm<<<gproj, 256, 0, stream>>>(ctxb, wpb, bpost, MTOT, D_MODEL, D_MODEL, hbuf, nullptr, inputs, 4);

  // LayerNorm(h) -> bf16
  k_layernorm<<<MTOT, 256, 0, stream>>>(hbuf, ln_g, ln_b, lnb);

  // FFN
  dim3 gffn1(HIDDEN / 128, MTOT / 128);    // (32, 64)
  k_gemm<<<gffn1, 256, 0, stream>>>(lnb, w1b, b1, MTOT, HIDDEN, D_MODEL, nullptr, f1b, nullptr, 2);
  dim3 gffn2(D_MODEL / 128, MTOT / 128);   // (8, 64)
  k_gemm<<<gffn2, 256, 0, stream>>>(f1b, w2b, b2, MTOT, D_MODEL, HIDDEN, out, nullptr, hbuf, 4);
}
